// EncoderDecoderAttentionGqa_9586367005398
// MI455X (gfx1250) — matrix operand order repaired and checked by reading, not compiled, not run on hardware
//
#include <hip/hip_runtime.h>
#include <hip/hip_bf16.h>
#include <stdint.h>

#define H 2048
#define NH 32
#define KVH 8
#define HD 64
#define GROUPS 4
#define BB 4
#define LL 1024
#define SS 1024
#define EPSL 1e-6f

typedef __bf16 v16bf __attribute__((ext_vector_type(16)));
typedef __bf16 v2bf  __attribute__((ext_vector_type(2)));
typedef float  v8f   __attribute__((ext_vector_type(8)));
typedef unsigned v4u __attribute__((ext_vector_type(4)));
typedef int      v4i __attribute__((ext_vector_type(4)));

__device__ __forceinline__ __bf16 f2bf(float f) { return (__bf16)f; }
// Native packed convert: matches v_cvt_pk_bf16_f32.
__device__ __forceinline__ unsigned pack2(float a, float b) {
  v2bf r;
  r.x = (__bf16)a;
  r.y = (__bf16)b;
  return __builtin_bit_cast(unsigned, r);
}
// Load a 16-element bf16 fragment as two 16-byte chunks (ISA frag layouts).
__device__ __forceinline__ v16bf load_frag(const __bf16* p0, const __bf16* p1) {
  union { v4u u[2]; v16bf v; } c;
  c.u[0] = *(const v4u*)p0;
  c.u[1] = *(const v4u*)p1;
  return c.v;
}

// gfx1250 async global->LDS copy (ASYNCcnt path), with safe fallback.
// Builtin prototype (probe-confirmed): (v4i addrspace(1)*, v4i addrspace(3)*, imm, imm)
__device__ __forceinline__ void async_copy16(const __bf16* g, __bf16* l) {
#if __has_builtin(__builtin_amdgcn_global_load_async_to_lds_b128)
  __builtin_amdgcn_global_load_async_to_lds_b128(
      (__attribute__((address_space(1))) v4i*)g,
      (__attribute__((address_space(3))) v4i*)l, 0, 0);
#else
  *(v4u*)l = *(const v4u*)g;
#endif
}
__device__ __forceinline__ void wait_async0() {
#if __has_builtin(__builtin_amdgcn_s_wait_asynccnt)
  __builtin_amdgcn_s_wait_asynccnt(0);
#else
  asm volatile("s_wait_asynccnt 0x0" ::: "memory");
#endif
}

// ---------------------------------------------------------------------------
// GEMM: Og[(b*heads+h)*L + l][HD] (bf16) = A(f32 MxK=2048) @ W(f32 KxN) + bias
// Block tile 64x256, K-chunk 32, 8 waves (2x4); wave tile 32x64 = 8 WMMAs
// from 2 A-frags x 4 B-frags per K-step.
// ---------------------------------------------------------------------------
__global__ __launch_bounds__(256) void gemm_qkv(
    const float* __restrict__ Ag, const float* __restrict__ Wg,
    const float* __restrict__ bias, __bf16* __restrict__ Og, int N) {
  __shared__ __align__(16) __bf16 As[64][32];
  __shared__ __align__(16) __bf16 Bs[256][32];   // W transposed: [n][k]
  const int t = threadIdx.x;
  const int lane = t & 31, wid = t >> 5;
  const int lo = lane & 15, hi = lane >> 4;
  const int wm = wid >> 2, wn = wid & 3;
  const int Mbase = blockIdx.y * 64;
  const int Nbase = blockIdx.x * 256;

  v8f c[2][4] = {};

  for (int kc = 0; kc < H; kc += 32) {
    __syncthreads();
#pragma unroll
    for (int i = 0; i < 2; ++i) {               // A tile 64x32 f32 -> bf16
      int idx = t + i * 256;                    // 512 float4
      int r = idx >> 3, c4 = (idx & 7) * 4;
      const float* ap = Ag + (size_t)(Mbase + r) * H + kc + c4;
      float4 v = *(const float4*)ap;
      __builtin_prefetch(ap + 32, 0, 0);        // speculative, no guard
      *(unsigned*)&As[r][c4]     = pack2(v.x, v.y);
      *(unsigned*)&As[r][c4 + 2] = pack2(v.z, v.w);
    }
#pragma unroll
    for (int i = 0; i < 8; ++i) {               // W tile 32x256, store [n][k]
      int idx = t + i * 256;                    // 2048 float4
      int kr = idx >> 6, nc4 = (idx & 63) * 4;
      const float* wp = Wg + (size_t)(kc + kr) * N + Nbase + nc4;
      float4 v = *(const float4*)wp;
      __builtin_prefetch(wp + (size_t)32 * N, 0, 0);
      Bs[nc4 + 0][kr] = f2bf(v.x);
      Bs[nc4 + 1][kr] = f2bf(v.y);
      Bs[nc4 + 2][kr] = f2bf(v.z);
      Bs[nc4 + 3][kr] = f2bf(v.w);
    }
    __syncthreads();
    v16bf a0 = load_frag(&As[wm * 32 + lo][hi * 8],      &As[wm * 32 + lo][hi * 8 + 16]);
    v16bf a1 = load_frag(&As[wm * 32 + 16 + lo][hi * 8], &As[wm * 32 + 16 + lo][hi * 8 + 16]);
#pragma unroll
    for (int nb = 0; nb < 4; ++nb) {
      const __bf16* bp = &Bs[wn * 64 + nb * 16 + lo][hi * 8];
      v16bf bf = load_frag(bp, bp + 16);
      c[0][nb] = __builtin_amdgcn_wmma_f32_16x16x32_bf16(false, a0, false, bf, (short)0, c[0][nb], false, false);
      c[1][nb] = __builtin_amdgcn_wmma_f32_16x16x32_bf16(false, a1, false, bf, (short)0, c[1][nb], false, false);
    }
  }

  const int heads = N >> 6;
#pragma unroll
  for (int nb = 0; nb < 4; ++nb) {
    int ng = Nbase + wn * 64 + nb * 16 + lo;
    float bvn = bias[ng];
    int hh = ng >> 6, d = ng & 63;
#pragma unroll
    for (int ma = 0; ma < 2; ++ma) {
#pragma unroll
      for (int j = 0; j < 8; ++j) {
        int mg = Mbase + wm * 32 + ma * 16 + hi * 8 + j;
        int bi = mg >> 10, li = mg & 1023;
        Og[(((size_t)(bi * heads + hh)) * LL + li) * HD + d] = f2bf(c[ma][nb][j] + bvn);
      }
    }
  }
}

// ---------------------------------------------------------------------------
// Flash attention: block = 4 waves = 64 q rows of one (b, head).
// ---------------------------------------------------------------------------
__global__ __launch_bounds__(128) void attn_kernel(
    const __bf16* __restrict__ Q, const __bf16* __restrict__ K,
    const __bf16* __restrict__ V, const float* __restrict__ mask,
    __bf16* __restrict__ ctx) {
  __shared__ __align__(16) __bf16 Vt[64][40];       // [d][s-chunk], padded
  __shared__ __align__(16) __bf16 Plds[4][16][32];

  const int t = threadIdx.x;
  const int lane = t & 31, wid = t >> 5;
  const int lo = lane & 15, hi = lane >> 4;

  const int qt = blockIdx.x & 15;                    // L/64 tiles
  const int bh = blockIdx.x >> 4;
  const int b = bh >> 5, h = bh & 31;
  const int kv = h >> 2;

  const __bf16* Qh = Q + ((size_t)(b * NH + h)) * LL * HD;
  const __bf16* Kh = K + ((size_t)(b * KVH + kv)) * SS * HD;
  const __bf16* Vh = V + ((size_t)(b * KVH + kv)) * SS * HD;
  const float*  mbase = mask + (size_t)b * LL * SS;

  const int q0 = qt * 64 + wid * 16;

  const __bf16* qp = Qh + (size_t)(q0 + lo) * HD;
  v16bf aq0 = load_frag(qp + hi * 8,      qp + hi * 8 + 16);       // k 0..31
  v16bf aq1 = load_frag(qp + 32 + hi * 8, qp + 32 + hi * 8 + 16);  // k 32..63

  v8f acc[4] = {v8f{}, v8f{}, v8f{}, v8f{}};
  float mrow[8], lrow[8];
#pragma unroll
  for (int j = 0; j < 8; ++j) { mrow[j] = -__builtin_inff(); lrow[j] = 0.f; }
  const float scale = 0.125f;                        // 1/sqrt(64)

  for (int s0 = 0; s0 < SS; s0 += 32) {
    __syncthreads();                                 // prev chunk consumed
    {                                                // stage V chunk transposed
      int r = t >> 2;                                // 0..31 (s)
      int dg = (t & 3) * 16;                         // d group
      const __bf16* vp = Vh + (size_t)(s0 + r) * HD + dg;
      __builtin_prefetch(vp + 32 * HD, 0, 0);        // speculative
#pragma unroll
      for (int i = 0; i < 16; ++i) Vt[dg + i][r] = vp[i];
    }
    __syncthreads();

    // ---- scores for two 16-key tiles
    v8f sc0 = {}, sc1 = {};
    {
      const __bf16* kp0 = Kh + (size_t)(s0 + lo) * HD;
      const __bf16* kp1 = Kh + (size_t)(s0 + 16 + lo) * HD;
      v16bf b00 = load_frag(kp0 + hi * 8,      kp0 + hi * 8 + 16);
      v16bf b01 = load_frag(kp0 + 32 + hi * 8, kp0 + 32 + hi * 8 + 16);
      v16bf b10 = load_frag(kp1 + hi * 8,      kp1 + hi * 8 + 16);
      v16bf b11 = load_frag(kp1 + 32 + hi * 8, kp1 + 32 + hi * 8 + 16);
      sc0 = __builtin_amdgcn_wmma_f32_16x16x32_bf16(false, aq0, false, b00, (short)0, sc0, false, false);
      sc0 = __builtin_amdgcn_wmma_f32_16x16x32_bf16(false, aq1, false, b01, (short)0, sc0, false, false);
      sc1 = __builtin_amdgcn_wmma_f32_16x16x32_bf16(false, aq0, false, b10, (short)0, sc1, false, false);
      sc1 = __builtin_amdgcn_wmma_f32_16x16x32_bf16(false, aq1, false, b11, (short)0, sc1, false, false);
    }

    // ---- online softmax (rows live across the 16-lane half-groups)
#pragma unroll
    for (int j = 0; j < 8; ++j) {
      int qr = q0 + hi * 8 + j;
      const float* mk = mbase + (size_t)qr * SS + s0;
      float x0 = sc0[j] * scale + mk[lo];
      float x1 = sc1[j] * scale + mk[16 + lo];
      float mx = fmaxf(x0, x1);
#pragma unroll
      for (int off = 1; off < 16; off <<= 1) mx = fmaxf(mx, __shfl_xor(mx, off, 32));
      float mnew = fmaxf(mrow[j], mx);
      float corr = __expf(mrow[j] - mnew);
      mrow[j] = mnew;
      float p0 = __expf(x0 - mnew);
      float p1 = __expf(x1 - mnew);
      float ps = p0 + p1;
#pragma unroll
      for (int off = 1; off < 16; off <<= 1) ps += __shfl_xor(ps, off, 32);
      lrow[j] = lrow[j] * corr + ps;
      acc[0][j] *= corr; acc[1][j] *= corr; acc[2][j] *= corr; acc[3][j] *= corr;
      Plds[wid][hi * 8 + j][lo]      = f2bf(p0);
      Plds[wid][hi * 8 + j][16 + lo] = f2bf(p1);
    }

    // ---- PV: out(16x64) += P(16x32) @ V(32x64)
    {
      const __bf16* pp = &Plds[wid][lo][0];
      v16bf ap = load_frag(pp + hi * 8, pp + hi * 8 + 16);
#pragma unroll
      for (int dt = 0; dt < 4; ++dt) {
        const __bf16* vtp = &Vt[dt * 16 + lo][hi * 8];
        v16bf bfr = load_frag(vtp, vtp + 16);
        acc[dt] = __builtin_amdgcn_wmma_f32_16x16x32_bf16(false, ap, false, bfr, (short)0, acc[dt], false, false);
      }
    }
  }

  // ---- normalize and store context bf16 at [b*L + l][h*HD + d]
#pragma unroll
  for (int j = 0; j < 8; ++j) {
    int qr = q0 + hi * 8 + j;
    float inv = 1.0f / lrow[j];
    size_t base = ((size_t)(b * LL + qr)) * H + h * HD + lo;
    ctx[base + 0]  = f2bf(acc[0][j] * inv);
    ctx[base + 16] = f2bf(acc[1][j] * inv);
    ctx[base + 32] = f2bf(acc[2][j] * inv);
    ctx[base + 48] = f2bf(acc[3][j] * inv);
  }
}

// ---------------------------------------------------------------------------
// Output projection: out(f32) = ctx(bf16) @ Wo + bo + residual
// Same 64x256 block / 32x64 wave tiling; A tile staged via async LDS DMA.
// ---------------------------------------------------------------------------
__global__ __launch_bounds__(256) void gemm_out(
    const __bf16* __restrict__ Ag, const float* __restrict__ Wg,
    const float* __restrict__ bias, const float* __restrict__ resid,
    float* __restrict__ Og) {
  __shared__ __align__(16) __bf16 As[64][32];
  __shared__ __align__(16) __bf16 Bs[256][32];
  const int t = threadIdx.x;
  const int lane = t & 31, wid = t >> 5;
  const int lo = lane & 15, hi = lane >> 4;
  const int wm = wid >> 2, wn = wid & 3;
  const int Mbase = blockIdx.y * 64;
  const int Nbase = blockIdx.x * 256;

  v8f c[2][4] = {};

  for (int kc = 0; kc < H; kc += 32) {
    __syncthreads();
    {                                             // A tile 64x32 bf16: async DMA
      int r = t >> 2, cc = (t & 3) * 8;
      const __bf16* ap = Ag + (size_t)(Mbase + r) * H + kc + cc;
      __builtin_prefetch(ap + 32, 0, 0);
      async_copy16(ap, &As[r][cc]);
    }
#pragma unroll
    for (int i = 0; i < 8; ++i) {                 // Wo tile 32x256 -> [n][k]
      int idx = t + i * 256;
      int kr = idx >> 6, nc4 = (idx & 63) * 4;
      const float* wp = Wg + (size_t)(kc + kr) * H + Nbase + nc4;
      float4 v = *(const float4*)wp;
      __builtin_prefetch(wp + (size_t)32 * H, 0, 0);
      Bs[nc4 + 0][kr] = f2bf(v.x);
      Bs[nc4 + 1][kr] = f2bf(v.y);
      Bs[nc4 + 2][kr] = f2bf(v.z);
      Bs[nc4 + 3][kr] = f2bf(v.w);
    }
    wait_async0();                                // A tile in LDS
    __syncthreads();
    v16bf a0 = load_frag(&As[wm * 32 + lo][hi * 8],      &As[wm * 32 + lo][hi * 8 + 16]);
    v16bf a1 = load_frag(&As[wm * 32 + 16 + lo][hi * 8], &As[wm * 32 + 16 + lo][hi * 8 + 16]);
#pragma unroll
    for (int nb = 0; nb < 4; ++nb) {
      const __bf16* bp = &Bs[wn * 64 + nb * 16 + lo][hi * 8];
      v16bf bf = load_frag(bp, bp + 16);
      c[0][nb] = __builtin_amdgcn_wmma_f32_16x16x32_bf16(false, a0, false, bf, (short)0, c[0][nb], false, false);
      c[1][nb] = __builtin_amdgcn_wmma_f32_16x16x32_bf16(false, a1, false, bf, (short)0, c[1][nb], false, false);
    }
  }

#pragma unroll
  for (int nb = 0; nb < 4; ++nb) {
    int ng = Nbase + wn * 64 + nb * 16 + lo;
    float bvn = bias[ng];
#pragma unroll
    for (int ma = 0; ma < 2; ++ma) {
#pragma unroll
      for (int j = 0; j < 8; ++j) {
        int mg = Mbase + wm * 32 + ma * 16 + hi * 8 + j;
        size_t ix = (size_t)mg * H + ng;
        Og[ix] = c[ma][nb][j] + bvn + resid[ix];
      }
    }
  }
}

// ---------------------------------------------------------------------------
// In-place LayerNorm over last dim (H=2048), one block per row.
// ---------------------------------------------------------------------------
__global__ __launch_bounds__(256) void ln_kernel(
    float* __restrict__ X, const float* __restrict__ gamma,
    const float* __restrict__ beta) {
  __shared__ float2 red[256];
  const int t = threadIdx.x;
  float* x = X + (size_t)blockIdx.x * H;
  float vals[8];
  float s = 0.f, s2 = 0.f;
#pragma unroll
  for (int i = 0; i < 8; ++i) {
    float v = x[t + i * 256];
    vals[i] = v; s += v; s2 += v * v;
  }
  red[t] = make_float2(s, s2);
  __syncthreads();
  for (int off = 128; off > 0; off >>= 1) {
    if (t < off) { red[t].x += red[t + off].x; red[t].y += red[t + off].y; }
    __syncthreads();
  }
  float mean = red[0].x * (1.0f / H);
  float var  = red[0].y * (1.0f / H) - mean * mean;
  float inv  = rsqrtf(var + EPSL);
#pragma unroll
  for (int i = 0; i < 8; ++i) {
    int col = t + i * 256;
    x[col] = (vals[i] - mean) * inv * gamma[col] + beta[col];
  }
}

extern "C" void kernel_launch(void* const* d_in, const int* in_sizes, int n_in,
                              void* d_out, int out_size, void* d_ws, size_t ws_size,
                              hipStream_t stream) {
  const float* hs  = (const float*)d_in[0];
  const float* ehs = (const float*)d_in[1];
  const float* msk = (const float*)d_in[2];
  const float* Wq  = (const float*)d_in[3];
  const float* bq  = (const float*)d_in[4];
  const float* Wk  = (const float*)d_in[5];
  const float* bk  = (const float*)d_in[6];
  const float* Wv  = (const float*)d_in[7];
  const float* bv  = (const float*)d_in[8];
  const float* Wo  = (const float*)d_in[9];
  const float* bo  = (const float*)d_in[10];
  const float* gm  = (const float*)d_in[11];
  const float* bt  = (const float*)d_in[12];
  float* out = (float*)d_out;

  char* ws = (char*)d_ws;
  __bf16* Qb = (__bf16*)(ws);                     // 16 MB: [b*NH+h][L][64]
  __bf16* Kb = (__bf16*)(ws + (16u << 20));       //  4 MB: [b*KVH+kv][S][64]
  __bf16* Vb = (__bf16*)(ws + (20u << 20));       //  4 MB
  __bf16* Cb = (__bf16*)(ws + (24u << 20));       // 16 MB: [b*L+l][H]

  dim3 blk(256);
  gemm_qkv<<<dim3(H / 256, (BB * LL) / 64), blk, 0, stream>>>(hs, Wq, bq, Qb, H);
  gemm_qkv<<<dim3((KVH * HD) / 256, (BB * SS) / 64), blk, 0, stream>>>(ehs, Wk, bk, Kb, KVH * HD);
  gemm_qkv<<<dim3((KVH * HD) / 256, (BB * SS) / 64), blk, 0, stream>>>(ehs, Wv, bv, Vb, KVH * HD);
  attn_kernel<<<dim3(BB * NH * (LL / 64)), dim3(128), 0, stream>>>(Qb, Kb, Vb, msk, Cb);
  gemm_out<<<dim3(H / 256, (BB * LL) / 64), blk, 0, stream>>>(Cb, Wo, bo, hs, out);
  ln_kernel<<<dim3(BB * LL), dim3(256), 0, stream>>>(out, gm, bt);
}
